// Context_Gate_TransformerBlock_12128987644508
// MI455X (gfx1250) — matrix operand order repaired and checked by reading, not compiled, not run on hardware
//
#include <hip/hip_runtime.h>

// ---------------------------------------------------------------------------
// Context_Gate_TransformerBlock for MI455X (gfx1250, wave32, WMMA bf16)
// ---------------------------------------------------------------------------
#define BATCH 8
#define DIM   192
#define HEADS 4
#define HD    48
#define HH    128
#define WW    128
#define HWPX  16384          // 128*128
#define PTOT  131072         // 8*16384
#define CQ    576            // 3*DIM
#define FFN   1020
#define FFNP  1024
#define HID   510
#define HIDP  512
#define CTX   256

typedef __attribute__((ext_vector_type(8)))  float   v8f;
typedef __attribute__((ext_vector_type(8)))  __bf16  v8bf;
typedef __attribute__((ext_vector_type(16))) __bf16  v16bf;

// A-tile (16x32, M x K) per-lane load: lanes 0-15 hold K {0..15},
// lanes 16-31 hold K {16..31}.  Caller pre-adds klo (0 or 16).
__device__ __forceinline__ v16bf ld_a(const __bf16* p) {
    v8bf lo = *(const v8bf*)(p);
    v8bf hi = *(const v8bf*)(p + 8);
    return __builtin_shufflevector(lo, hi, 0,1,2,3,4,5,6,7,8,9,10,11,12,13,14,15);
}
// B-tile (32x16, K x N) per-lane load: contiguous 16 K values (caller adds hi*16).
__device__ __forceinline__ v16bf ld_b(const __bf16* p) {
    return *(const v16bf*)(p);
}
__device__ __forceinline__ v8f wmma_bf16(v16bf a, v16bf b, v8f c) {
    return __builtin_amdgcn_wmma_f32_16x16x32_bf16(false, a, false, b, (short)0, c, false, false);
}
__device__ __forceinline__ v8f v8f_zero() {
    v8f z;
#pragma unroll
    for (int i = 0; i < 8; ++i) z[i] = 0.f;
    return z;
}

// ---------------------------------------------------------------------------
// Weight fp32 -> bf16 with optional row/col zero padding. dst: (rdst, kdst)
// ---------------------------------------------------------------------------
__global__ void wconv_kernel(__bf16* __restrict__ dst, const float* __restrict__ src,
                             int rdst, int kdst, int rsrc, int ksrc) {
    int i = blockIdx.x * 256 + threadIdx.x;
    if (i >= rdst * kdst) return;
    int r = i / kdst, k = i - r * kdst;
    float v = (r < rsrc && k < ksrc) ? src[r * ksrc + k] : 0.f;
    dst[i] = (__bf16)v;
}

// ---------------------------------------------------------------------------
// Context adapters: temperature (B,HEADS) and v-gate (B,DIM)
// ---------------------------------------------------------------------------
__global__ void adapters_kernel(const float* __restrict__ ctx,
                                const float* __restrict__ taw1, const float* __restrict__ tab1,
                                const float* __restrict__ taw2, const float* __restrict__ tab2,
                                const float* __restrict__ btemp,
                                const float* __restrict__ vgw,  const float* __restrict__ vgb,
                                float* __restrict__ tempo, float* __restrict__ vgate) {
    __shared__ float hid[48];
    int b = blockIdx.x, t = threadIdx.x;
    if (t < 48) {
        float s = tab1[t];
        for (int k = 0; k < CTX; ++k) s += ctx[b * CTX + k] * taw1[t * CTX + k];
        hid[t] = fmaxf(s, 0.f);
    }
    __syncthreads();
    if (t < HEADS) {
        float s = tab2[t];
        for (int j = 0; j < 48; ++j) s += hid[j] * taw2[t * 48 + j];
        float sg = 1.f / (1.f + expf(-s));
        tempo[b * HEADS + t] = btemp[t] * (sg * 2.f + 0.5f);
    }
    if (t < DIM) {
        float s = vgb[t];
        for (int k = 0; k < CTX; ++k) s += ctx[b * CTX + k] * vgw[t * CTX + k];
        vgate[b * DIM + t] = 1.f / (1.f + expf(-s));
    }
}

// ---------------------------------------------------------------------------
// LayerNorm over channels.  NCHW=true: input is (B,C,H,W) fp32 (transposes to
// pixel-major).  NCHW=false: input is (P,C) fp32.  Output (P,C) bf16.
// Block: 256 threads, 32 pixels x 192 channels tile.
// ---------------------------------------------------------------------------
template<bool NCHW>
__global__ __launch_bounds__(256)
void ln_kernel(const float* __restrict__ in, const float* __restrict__ w,
               const float* __restrict__ bia, __bf16* __restrict__ out) {
    __shared__ float tile[DIM * 32];
    __shared__ float ps1[8 * 32], ps2[8 * 32];
    __shared__ float mu_s[32], rs_s[32];
    int t = threadIdx.x;
    size_t p0 = (size_t)blockIdx.x * 32;
    int b = (int)(p0 >> 14), s0 = (int)(p0 & 16383);
    for (int i = t; i < DIM * 32; i += 256) {
        int c = i >> 5, px = i & 31;
        float v;
        if (NCHW) v = in[(((size_t)(b * DIM + c)) << 14) + s0 + px];
        else      v = in[(p0 + px) * DIM + c];
        tile[c * 32 + px] = v;
    }
    __syncthreads();
    {
        int px = t & 31, g = t >> 5;
        float s1 = 0.f, s2 = 0.f;
        for (int c = g * 24; c < g * 24 + 24; ++c) {
            float v = tile[c * 32 + px]; s1 += v; s2 += v * v;
        }
        ps1[g * 32 + px] = s1; ps2[g * 32 + px] = s2;
    }
    __syncthreads();
    if (t < 32) {
        float s1 = 0.f, s2 = 0.f;
        for (int g = 0; g < 8; ++g) { s1 += ps1[g * 32 + t]; s2 += ps2[g * 32 + t]; }
        float mu  = s1 * (1.f / DIM);
        float var = s2 * (1.f / DIM) - mu * mu;
        mu_s[t] = mu; rs_s[t] = rsqrtf(var + 1e-5f);
    }
    __syncthreads();
    for (int i = t; i < DIM * 32; i += 256) {
        int px = i / DIM, c = i - px * DIM;
        float v = (tile[c * 32 + px] - mu_s[px]) * rs_s[px] * w[c] + bia[c];
        out[(p0 + px) * DIM + c] = (__bf16)v;
    }
}

// ---------------------------------------------------------------------------
// Main GEMM:  out[p,n] = sum_k A[p,k] * Bw[n,k]   (A pixel-major, Bw (N,Kpad))
// Block 256 = 8 waves; each wave: 32(M) x 64(N) register tile, WMMA bf16.
// EPI 0: store bf16 (row len Nout)
// EPI 1: += x (NCHW residual), store f32 pixel-major (proj)
// EPI 2: += x1 (pixel-major residual), store f32 NCHW (final output)
// ---------------------------------------------------------------------------
template<int EPI>
__global__ __launch_bounds__(256)
void gemm_kernel(const __bf16* __restrict__ A, const __bf16* __restrict__ Bw,
                 void* __restrict__ outp, const float* __restrict__ resid,
                 int Kpad, int Nout) {
    const int lane = threadIdx.x & 31, wave = threadIdx.x >> 5;
    const int lm = lane & 15, hi = lane >> 4;
    const int klo = hi ? 16 : 0;
    const size_t pBase = (size_t)blockIdx.x * 256 + wave * 32;
    const int nBase = blockIdx.y * 64;

    const __bf16* a0p = A + (pBase + lm)      * (size_t)Kpad + klo;
    const __bf16* a1p = A + (pBase + 16 + lm) * (size_t)Kpad + klo;
    const __bf16* bp[4];
#pragma unroll
    for (int j = 0; j < 4; ++j)
        bp[j] = Bw + (size_t)(nBase + j * 16 + lm) * Kpad + hi * 16;

    v8f acc[2][4];
#pragma unroll
    for (int i = 0; i < 2; ++i)
#pragma unroll
        for (int j = 0; j < 4; ++j) acc[i][j] = v8f_zero();

    const int ksteps = Kpad >> 5;
    for (int ks = 0; ks < ksteps; ++ks) {
        v16bf a0 = ld_a(a0p + ks * 32);
        v16bf a1 = ld_a(a1p + ks * 32);
#pragma unroll
        for (int j = 0; j < 4; ++j) {
            v16bf b = ld_b(bp[j] + ks * 32);
            acc[0][j] = wmma_bf16(a0, b, acc[0][j]);
            acc[1][j] = wmma_bf16(a1, b, acc[1][j]);
        }
    }
#pragma unroll
    for (int i = 0; i < 2; ++i)
#pragma unroll
        for (int j = 0; j < 4; ++j)
#pragma unroll
            for (int r = 0; r < 8; ++r) {
                size_t p = pBase + i * 16 + r + hi * 8;
                int n = nBase + j * 16 + lm;
                float v = acc[i][j][r];
                if (EPI == 0) {
                    ((__bf16*)outp)[p * Nout + n] = (__bf16)v;
                } else if (EPI == 1) {
                    size_t bb = p >> 14, s = p & 16383;
                    v += resid[((bb * DIM + n) << 14) + s];
                    ((float*)outp)[p * DIM + n] = v;
                } else {
                    v += resid[p * DIM + n];
                    size_t bb = p >> 14, s = p & 16383;
                    ((float*)outp)[((bb * DIM + n) << 14) + s] = v;
                }
            }
}

// ---------------------------------------------------------------------------
// Depthwise 3x3 (SAME, zero pad), channels-last.  Optional fused add.
// ---------------------------------------------------------------------------
template<bool ADD>
__global__ void dwconv_kernel(const __bf16* __restrict__ in, int ldin, int coff,
                              const float* __restrict__ wdw, int C,
                              const __bf16* __restrict__ addp, int ldadd,
                              __bf16* __restrict__ out, int ldout) {
    int c = blockIdx.y * 256 + threadIdx.x;
    if (c >= C) return;
    size_t p = blockIdx.x;
    int s = (int)(p & 16383), y = s >> 7, x = s & 127;
    const float* wc = wdw + c * 9;
    float acc = 0.f;
#pragma unroll
    for (int dy = -1; dy <= 1; ++dy) {
        int yy = y + dy; if (yy < 0 || yy > 127) continue;
#pragma unroll
        for (int dx = -1; dx <= 1; ++dx) {
            int xx = x + dx; if (xx < 0 || xx > 127) continue;
            size_t pn = p + (size_t)(dy * 128 + dx);
            acc += (float)in[pn * ldin + coff + c] * wc[(dy + 1) * 3 + (dx + 1)];
        }
    }
    if (ADD) acc += (float)addp[p * ldadd + c];
    out[p * ldout + c] = (__bf16)acc;
}

// ---------------------------------------------------------------------------
// FFN depthwise + exact-GELU gating: g[p,c] = gelu(dw(c)) * dw(c+510)
// ---------------------------------------------------------------------------
__global__ void ffn_dwgate_kernel(const __bf16* __restrict__ yp,
                                  const float* __restrict__ wdw,
                                  __bf16* __restrict__ g) {
    int c = blockIdx.y * 256 + threadIdx.x;          // 0..511
    size_t p = blockIdx.x;
    if (c >= HID) { if (c < HIDP) g[p * HIDP + c] = (__bf16)0.f; return; }
    int s = (int)(p & 16383), y = s >> 7, x = s & 127;
    const float* w1 = wdw + c * 9;
    const float* w2 = wdw + (c + HID) * 9;
    float a1 = 0.f, a2 = 0.f;
#pragma unroll
    for (int dy = -1; dy <= 1; ++dy) {
        int yy = y + dy; if (yy < 0 || yy > 127) continue;
#pragma unroll
        for (int dx = -1; dx <= 1; ++dx) {
            int xx = x + dx; if (xx < 0 || xx > 127) continue;
            size_t pn = p + (size_t)(dy * 128 + dx);
            float wgt1 = w1[(dy + 1) * 3 + (dx + 1)];
            float wgt2 = w2[(dy + 1) * 3 + (dx + 1)];
            a1 += (float)yp[pn * FFNP + c]       * wgt1;
            a2 += (float)yp[pn * FFNP + c + HID] * wgt2;
        }
    }
    float ge = 0.5f * a1 * (1.f + erff(a1 * 0.70710678118654752f));
    g[p * HIDP + c] = (__bf16)(ge * a2);
}

// ---------------------------------------------------------------------------
// q/k spatial L2-normalize + transpose to channel-major (b, c, 16384) bf16.
// One block per (b, qk-channel 0..383).
// ---------------------------------------------------------------------------
__global__ void qknorm_kernel(const __bf16* __restrict__ qkv,
                              __bf16* __restrict__ qnT, __bf16* __restrict__ knT) {
    __shared__ float red[256];
    int t = threadIdx.x;
    int idx = blockIdx.x;                  // 0..3071
    int b = idx / 384, ch = idx - b * 384; // qkv channel == ch (q:0..191, k:192..383)
    const __bf16* src = qkv + (size_t)b * HWPX * CQ + ch;
    float ss = 0.f;
    for (int s = t; s < HWPX; s += 256) { float v = (float)src[(size_t)s * CQ]; ss += v * v; }
    red[t] = ss;
    __syncthreads();
    for (int off = 128; off > 0; off >>= 1) {
        if (t < off) red[t] += red[t + off];
        __syncthreads();
    }
    float rn = 1.f / fmaxf(sqrtf(red[0]), 1e-12f);
    __bf16* dst = (ch < DIM) ? (qnT + ((size_t)(b * DIM + ch)) * HWPX)
                             : (knT + ((size_t)(b * DIM + ch - DIM)) * HWPX);
    for (int s = t; s < HWPX; s += 256) dst[s] = (__bf16)((float)src[(size_t)s * CQ] * rn);
}

// ---------------------------------------------------------------------------
// Per-(b,h): logits = scale*temp * (qn^T kn) via WMMA (K=16384 split over 8
// waves, LDS ds_add_f32 reduce) -> softmax -> * v_gate -> attn_g (48 x 64,
// n-major, K-padded with zeros) bf16.
// ---------------------------------------------------------------------------
__global__ __launch_bounds__(256)
void attn_kernel(const __bf16* __restrict__ qnT, const __bf16* __restrict__ knT,
                 const float* __restrict__ tempo, const float* __restrict__ vgate,
                 __bf16* __restrict__ attn_g) {
    __shared__ float slog[HD * HD];
    __shared__ float srow[HD];
    int t = threadIdx.x, lane = t & 31, wave = t >> 5;
    int bh = blockIdx.x, b = bh >> 2, h = bh & 3;
    int lm = lane & 15, hi = lane >> 4, klo = hi ? 16 : 0;

    const __bf16* qb = qnT + ((size_t)(b * DIM + h * HD)) * HWPX;
    const __bf16* kb = knT + ((size_t)(b * DIM + h * HD)) * HWPX;
    const __bf16* ap[3]; const __bf16* bp[3];
#pragma unroll
    for (int i = 0; i < 3; ++i) ap[i] = qb + (size_t)(i * 16 + lm) * HWPX + klo;
#pragma unroll
    for (int j = 0; j < 3; ++j) bp[j] = kb + (size_t)(j * 16 + lm) * HWPX + hi * 16;

    v8f acc[3][3];
#pragma unroll
    for (int i = 0; i < 3; ++i)
#pragma unroll
        for (int j = 0; j < 3; ++j) acc[i][j] = v8f_zero();

    int k0 = wave * 2048;
    for (int ks = 0; ks < 64; ++ks) {
        int kk = k0 + ks * 32;
        v16bf a[3], bb[3];
#pragma unroll
        for (int i = 0; i < 3; ++i) a[i] = ld_a(ap[i] + kk);
#pragma unroll
        for (int j = 0; j < 3; ++j) bb[j] = ld_b(bp[j] + kk);
#pragma unroll
        for (int i = 0; i < 3; ++i)
#pragma unroll
            for (int j = 0; j < 3; ++j) acc[i][j] = wmma_bf16(a[i], bb[j], acc[i][j]);
    }
    for (int i = t; i < HD * HD; i += 256) slog[i] = 0.f;
    __syncthreads();
#pragma unroll
    for (int i = 0; i < 3; ++i)
#pragma unroll
        for (int j = 0; j < 3; ++j)
#pragma unroll
            for (int r = 0; r < 8; ++r) {
                int c = i * 16 + r + hi * 8;
                int d = j * 16 + lm;
                atomicAdd(&slog[c * HD + d], acc[i][j][r]);
            }
    __syncthreads();
    if (t < HD) {
        float tv = tempo[b * HEADS + h] * 0.14433756729740643f;  // scale = 48^-0.5
        float mx = -1e30f;
        for (int d = 0; d < HD; ++d) mx = fmaxf(mx, slog[t * HD + d] * tv);
        float sum = 0.f;
        for (int d = 0; d < HD; ++d) {
            float e = expf(slog[t * HD + d] * tv - mx);
            slog[t * HD + d] = e; sum += e;
        }
        srow[t] = 1.f / sum;
    }
    __syncthreads();
    for (int i = t; i < HD * 64; i += 256) {
        int c = i >> 6, d = i & 63;
        float v = 0.f;
        if (d < HD) v = slog[c * HD + d] * srow[c] * vgate[b * DIM + h * HD + d];
        attn_g[((size_t)(bh * HD + c)) * 64 + d] = (__bf16)v;
    }
}

// ---------------------------------------------------------------------------
// out_global[p, h*48+c] = sum_d v[p, 384+h*48+d] * attn_g[bh][c][d]  (WMMA,
// K=64 padded; A overrun into next row covered by slack + zero B rows).
// ---------------------------------------------------------------------------
__global__ __launch_bounds__(256)
void outg_kernel(const __bf16* __restrict__ qkv, const __bf16* __restrict__ attn_g,
                 __bf16* __restrict__ og) {
    int lane = threadIdx.x & 31, wave = threadIdx.x >> 5;
    int lm = lane & 15, hi = lane >> 4, klo = hi ? 16 : 0;
    int h = blockIdx.y;
    size_t pBase = (size_t)blockIdx.x * 256 + wave * 32;
    int b = (int)(pBase >> 14);
    const __bf16* a0p = qkv + (pBase + lm)      * CQ + 384 + h * HD + klo;
    const __bf16* a1p = qkv + (pBase + 16 + lm) * CQ + 384 + h * HD + klo;
    const __bf16* bp[3];
#pragma unroll
    for (int j = 0; j < 3; ++j)
        bp[j] = attn_g + ((size_t)((b * HEADS + h) * HD + j * 16 + lm)) * 64 + hi * 16;

    v8f acc[2][3];
#pragma unroll
    for (int i = 0; i < 2; ++i)
#pragma unroll
        for (int j = 0; j < 3; ++j) acc[i][j] = v8f_zero();

#pragma unroll
    for (int ks = 0; ks < 2; ++ks) {
        v16bf a0 = ld_a(a0p + ks * 32);
        v16bf a1 = ld_a(a1p + ks * 32);
#pragma unroll
        for (int j = 0; j < 3; ++j) {
            v16bf bb = ld_b(bp[j] + ks * 32);
            acc[0][j] = wmma_bf16(a0, bb, acc[0][j]);
            acc[1][j] = wmma_bf16(a1, bb, acc[1][j]);
        }
    }
#pragma unroll
    for (int i = 0; i < 2; ++i)
#pragma unroll
        for (int j = 0; j < 3; ++j)
#pragma unroll
            for (int r = 0; r < 8; ++r) {
                size_t p = pBase + i * 16 + r + hi * 8;
                int n = j * 16 + lm;
                og[p * DIM + h * HD + n] = (__bf16)acc[i][j][r];
            }
}

// ---------------------------------------------------------------------------
// Workspace arena (bytes, all offsets 256B aligned).  Peak ~722 MB.
// ---------------------------------------------------------------------------
static constexpr size_t O_WQ    = 0;                          // 576x192 bf16
static constexpr size_t O_WP    = 221184;                     // 192x192 bf16
static constexpr size_t O_WFI   = 294912;                     // 1024x192 bf16 (pad rows)
static constexpr size_t O_WFO   = 688128;                     // 192x512 bf16 (pad cols)
static constexpr size_t O_TEMP  = 1048576;                    // 8x4 f32
static constexpr size_t O_VGATE = 1048576 + 256;              // 8x192 f32
static constexpr size_t O_ATTG  = 1048576 + 8192;             // 32x48x64 bf16
static constexpr size_t O_XN1   = 2097152;                    // (P,192) bf16; reused: qnT
static constexpr size_t O_BUF2  = O_XN1 + 50331648;           // qkv_pre (P,576) bf16;
                                                              // reused: knT | og | comb; later xn2
static constexpr size_t O_QKV   = O_BUF2 + 150994944;         // qkv (P,576) bf16 (+4KB slack);
                                                              // reused: x1 (P,192) f32
static constexpr size_t O_YPRE  = O_QKV + 150994944 + 4096;   // (P,1024) bf16
static constexpr size_t O_G     = O_YPRE + 268435456;         // (P,512) bf16

extern "C" void kernel_launch(void* const* d_in, const int* in_sizes, int n_in,
                              void* d_out, int out_size, void* d_ws, size_t ws_size,
                              hipStream_t stream) {
    const float* x      = (const float*)d_in[0];
    const float* ctx    = (const float*)d_in[1];
    const float* ln1w   = (const float*)d_in[2];
    const float* ln1b   = (const float*)d_in[3];
    const float* ln2w   = (const float*)d_in[4];
    const float* ln2b   = (const float*)d_in[5];
    const float* wqkv   = (const float*)d_in[6];
    const float* wqkvdw = (const float*)d_in[7];
    const float* wproj  = (const float*)d_in[8];
    const float* btemp  = (const float*)d_in[9];
    const float* taw1   = (const float*)d_in[10];
    const float* tab1   = (const float*)d_in[11];
    const float* taw2   = (const float*)d_in[12];
    const float* tab2   = (const float*)d_in[13];
    const float* vgw    = (const float*)d_in[14];
    const float* vgb    = (const float*)d_in[15];
    const float* wlocal = (const float*)d_in[16];
    const float* wffi   = (const float*)d_in[17];
    const float* wffdw  = (const float*)d_in[18];
    const float* wffo   = (const float*)d_in[19];

    char* ws = (char*)d_ws;
    __bf16* Wq    = (__bf16*)(ws + O_WQ);
    __bf16* Wp    = (__bf16*)(ws + O_WP);
    __bf16* Wfi   = (__bf16*)(ws + O_WFI);
    __bf16* Wfo   = (__bf16*)(ws + O_WFO);
    float*  tempo = (float*) (ws + O_TEMP);
    float*  vgate = (float*) (ws + O_VGATE);
    __bf16* attng = (__bf16*)(ws + O_ATTG);
    __bf16* xn1   = (__bf16*)(ws + O_XN1);
    __bf16* qnT   = (__bf16*)(ws + O_XN1);                 // reuse (xn1 dead)
    __bf16* qkvp  = (__bf16*)(ws + O_BUF2);
    __bf16* knT   = (__bf16*)(ws + O_BUF2);                // reuse (qkv_pre dead)
    __bf16* og    = (__bf16*)(ws + O_BUF2 + 50331648);
    __bf16* comb  = (__bf16*)(ws + O_BUF2 + 100663296);
    __bf16* xn2   = (__bf16*)(ws + O_BUF2);                // reuse (attn buffers dead)
    __bf16* qkv   = (__bf16*)(ws + O_QKV);
    float*  x1    = (float*) (ws + O_QKV);                 // reuse (v dead after local dw)
    __bf16* ypre  = (__bf16*)(ws + O_YPRE);
    __bf16* g     = (__bf16*)(ws + O_G);
    float*  out   = (float*)d_out;

    // --- weight conversion / padding ---
    wconv_kernel<<<(CQ * DIM + 255) / 256, 256, 0, stream>>>(Wq, wqkv, CQ, DIM, CQ, DIM);
    wconv_kernel<<<(DIM * DIM + 255) / 256, 256, 0, stream>>>(Wp, wproj, DIM, DIM, DIM, DIM);
    wconv_kernel<<<(FFNP * DIM + 255) / 256, 256, 0, stream>>>(Wfi, wffi, FFNP, DIM, FFN, DIM);
    wconv_kernel<<<(DIM * HIDP + 255) / 256, 256, 0, stream>>>(Wfo, wffo, DIM, HIDP, DIM, HID);

    // --- context adapters ---
    adapters_kernel<<<BATCH, 256, 0, stream>>>(ctx, taw1, tab1, taw2, tab2, btemp,
                                               vgw, vgb, tempo, vgate);

    // --- attention branch ---
    ln_kernel<true><<<PTOT / 32, 256, 0, stream>>>(x, ln1w, ln1b, xn1);
    gemm_kernel<0><<<dim3(PTOT / 256, 9), 256, 0, stream>>>(xn1, Wq, qkvp, nullptr, DIM, CQ);
    dwconv_kernel<false><<<dim3(PTOT, 3), 256, 0, stream>>>(qkvp, CQ, 0, wqkvdw, CQ,
                                                            (const __bf16*)nullptr, 0, qkv, CQ);
    qknorm_kernel<<<BATCH * 2 * DIM, 256, 0, stream>>>(qkv, qnT, knT);
    attn_kernel<<<BATCH * HEADS, 256, 0, stream>>>(qnT, knT, tempo, vgate, attng);
    outg_kernel<<<dim3(PTOT / 256, HEADS), 256, 0, stream>>>(qkv, attng, og);
    dwconv_kernel<true><<<dim3(PTOT, 1), 256, 0, stream>>>(qkv, CQ, 384, wlocal, DIM,
                                                           og, DIM, comb, DIM);
    gemm_kernel<1><<<dim3(PTOT / 256, 3), 256, 0, stream>>>(comb, Wp, x1, x, DIM, DIM);

    // --- GDFN branch ---
    ln_kernel<false><<<PTOT / 32, 256, 0, stream>>>(x1, ln2w, ln2b, xn2);
    gemm_kernel<0><<<dim3(PTOT / 256, 16), 256, 0, stream>>>(xn2, Wfi, ypre, nullptr, DIM, FFNP);
    ffn_dwgate_kernel<<<dim3(PTOT, 2), 256, 0, stream>>>(ypre, wffdw, g);
    gemm_kernel<2><<<dim3(PTOT / 256, 3), 256, 0, stream>>>(g, Wfo, out, x1, HIDP, DIM);
}
